// TransformerBlock_38508676776335
// MI455X (gfx1250) — compile-verified, matrix operand order checked by reading, not run on hardware
//
#include <hip/hip_runtime.h>
#include <hip/hip_bf16.h>
#include <math.h>

// Problem constants (match reference)
#define B_  4
#define T_  2048
#define C_  1024
#define NH_ 16
#define HD_ 64
#define FF_ 4096
#define BT_ (B_ * T_)
#define BK_ 64   // GEMM K-step (two 16x16x32 WMMA K-chunks per step)

typedef __attribute__((ext_vector_type(8)))  float  v8f;
typedef __attribute__((ext_vector_type(16))) __bf16 v16bf;
typedef __attribute__((ext_vector_type(8)))  __bf16 v8bf;

union FragB16 {
  v16bf v;
  v8bf  h[2];
  uint4 q[2];
};

static __device__ __forceinline__ v8f wmma_bf16(v16bf a, v16bf b, v8f c) {
  // D = A(16x32 bf16) * B(32x16 bf16) + C(16x16 f32)
  return __builtin_amdgcn_wmma_f32_16x16x32_bf16(
      /*neg_a=*/false, a, /*neg_b=*/false, b,
      /*c_mod=*/(short)0, c, /*reuse_a=*/false, /*reuse_b=*/false);
}

static __device__ __forceinline__ float gelu_f(float x) {
  const float k = 0.7978845608028654f; // sqrt(2/pi)
  return 0.5f * x * (1.0f + tanhf(k * (x + 0.044715f * x * x * x)));
}

// LDS byte address of a __shared__ object (low 32 bits of the generic address)
static __device__ __forceinline__ uint32_t lds_lo(const void* p) {
  return (uint32_t)(uintptr_t)p;
}

// CDNA5 async memory->LDS DMA (tracked by ASYNCcnt), 16B per lane.
static __device__ __forceinline__ void async_ld_b128(uint32_t lds_addr, const void* gaddr) {
  asm volatile("global_load_async_to_lds_b128 %0, %1, off"
               :: "v"(lds_addr), "v"(gaddr)
               : "memory");
}
static __device__ __forceinline__ void wait_async0() {
  asm volatile("s_wait_asynccnt 0" ::: "memory");
}
static __device__ __forceinline__ void wait_ds0() {
  asm volatile("s_wait_dscnt 0" ::: "memory");
}

// ---------------------------------------------------------------------------
// Weight transpose + fp32 -> bf16 convert: w[K,N] -> wt[N,K]
// ---------------------------------------------------------------------------
__global__ __launch_bounds__(256) void transpose_convert_kernel(
    const float* __restrict__ w, __bf16* __restrict__ wt, int K, int N) {
  __shared__ float tile[32][33];
  const int n0 = blockIdx.x * 32, k0 = blockIdx.y * 32;
  const int tx = threadIdx.x, ty = threadIdx.y; // block (32,8)
#pragma unroll
  for (int i = 0; i < 4; ++i)
    tile[ty + 8 * i][tx] = w[(size_t)(k0 + ty + 8 * i) * N + (n0 + tx)];
  __syncthreads();
#pragma unroll
  for (int i = 0; i < 4; ++i)
    wt[(size_t)(n0 + ty + 8 * i) * K + (k0 + tx)] = (__bf16)tile[tx][ty + 8 * i];
}

// ---------------------------------------------------------------------------
// LayerNorm over last dim (C=1024), fp32 in -> bf16 out. 256 thr/row.
// ---------------------------------------------------------------------------
__global__ __launch_bounds__(256) void layernorm_bf16_kernel(
    const float* __restrict__ x, const float* __restrict__ g,
    const float* __restrict__ bta, __bf16* __restrict__ out) {
  const int row = blockIdx.x;
  const float* xr = x + (size_t)row * C_;
  float v[4], s = 0.f, ss = 0.f;
#pragma unroll
  for (int i = 0; i < 4; ++i) {
    v[i] = xr[threadIdx.x + 256 * i];
    s += v[i];
    ss += v[i] * v[i];
  }
#pragma unroll
  for (int off = 16; off > 0; off >>= 1) {
    s  += __shfl_xor(s, off, 32);
    ss += __shfl_xor(ss, off, 32);
  }
  __shared__ float red[16];
  const int wid = threadIdx.x >> 5, lane = threadIdx.x & 31;
  if (lane == 0) { red[wid] = s; red[wid + 8] = ss; }
  __syncthreads();
  if (threadIdx.x == 0) {
    float ts = 0.f, tss = 0.f;
#pragma unroll
    for (int i = 0; i < 8; ++i) { ts += red[i]; tss += red[i + 8]; }
    red[0] = ts; red[8] = tss;
  }
  __syncthreads();
  const float mu   = red[0] * (1.0f / C_);
  const float var  = red[8] * (1.0f / C_) - mu * mu;
  const float rstd = rsqrtf(var + 1e-5f);
#pragma unroll
  for (int i = 0; i < 4; ++i) {
    const int c = threadIdx.x + 256 * i;
    out[(size_t)row * C_ + c] = (__bf16)(((v[i] - mu) * rstd) * g[c] + bta[c]);
  }
}

// ---------------------------------------------------------------------------
// WMMA GEMM: out[M,N] = A[M,K](bf16) * Wt[N,K]^T(bf16) + bias (+gelu)(+resid)
// WG tile 64(M) x 128(N), 8 waves = 4(M) x 2(N). K-step 64.
// Double-buffered LDS fed by global_load_async_to_lds_b128 (ASYNCcnt):
// tile t+1 DMA overlaps tile t WMMA; one s_wait_asynccnt + barrier per tile.
// ---------------------------------------------------------------------------
__global__ __launch_bounds__(256) void gemm_bf16_kernel(
    const __bf16* __restrict__ A, const __bf16* __restrict__ Wt,
    const float* __restrict__ bias, const float* __restrict__ resid,
    float* __restrict__ out_f32, __bf16* __restrict__ out_bf16,
    int M, int N, int K, int do_gelu) {
  __shared__ __bf16 As[2][64 * BK_];   // 2 x 8 KB
  __shared__ __bf16 Bs[2][128 * BK_];  // 2 x 16 KB
  const int m0 = blockIdx.y * 64;
  const int n0 = blockIdx.x * 128;
  const int tid  = threadIdx.x;
  const int w    = tid >> 5;
  const int lane = tid & 31;
  const int lh = lane & 15;   // row/col within fragment
  const int hi = lane >> 4;   // K-half selector
  const int mw = w >> 1;      // 0..3 -> 16-row slice
  const int nw = w & 1;       // 0..1 -> 64-col slice

  v8f acc[4];
#pragma unroll
  for (int j = 0; j < 4; ++j)
#pragma unroll
    for (int e = 0; e < 8; ++e) acc[j][e] = 0.0f;

  // Staging assignment: A tile 64x64 (16 el/thr), B tile 128x64 (32 el/thr)
  const int arow = tid >> 2, ac = (tid & 3) * 16;
  const int brow = tid >> 1, bc = (tid & 1) * 32;
  const __bf16* Ag = A  + (size_t)(m0 + arow) * K + ac;
  const __bf16* Bg = Wt + (size_t)(n0 + brow) * K + bc;
  uint32_t asA[2], asB[2];
#pragma unroll
  for (int bfi = 0; bfi < 2; ++bfi) {
    asA[bfi] = lds_lo(&As[bfi][arow * BK_ + ac]);
    asB[bfi] = lds_lo(&Bs[bfi][brow * BK_ + bc]);
  }

  // issue one K-tile of async global->LDS DMA (6 x b128 per thread)
  auto issue = [&](int kk, int buf) {
    async_ld_b128(asA[buf],      Ag + kk);
    async_ld_b128(asA[buf] + 16, Ag + kk + 8);
    async_ld_b128(asB[buf],      Bg + kk);
    async_ld_b128(asB[buf] + 16, Bg + kk + 8);
    async_ld_b128(asB[buf] + 32, Bg + kk + 16);
    async_ld_b128(asB[buf] + 48, Bg + kk + 24);
  };

  const int ntiles = K / BK_;
  issue(0, 0);
  for (int t = 0; t < ntiles; ++t) {
    const int buf = t & 1;
    wait_async0();     // this wave's DMA for tile t landed in LDS
    __syncthreads();   // all waves' DMA landed; prior tile reads finished
    if (t + 1 < ntiles) issue((t + 1) * BK_, buf ^ 1);

    const __bf16* Ab = &As[buf][0];
    const __bf16* Bb = &Bs[buf][0];
    FragB16 a0, a1;
    const int arf = (mw * 16 + lh) * BK_;
    a0.q[0] = *(const uint4*)&Ab[arf + hi * 8];
    a0.q[1] = *(const uint4*)&Ab[arf + 16 + hi * 8];
    a1.q[0] = *(const uint4*)&Ab[arf + 32 + hi * 8];
    a1.q[1] = *(const uint4*)&Ab[arf + 48 + hi * 8];
#pragma unroll
    for (int j = 0; j < 4; ++j) {
      const int rn = (nw * 64 + j * 16 + lh) * BK_;
      FragB16 b0, b1;
      b0.q[0] = *(const uint4*)&Bb[rn + hi * 8];
      b0.q[1] = *(const uint4*)&Bb[rn + 16 + hi * 8];
      acc[j] = wmma_bf16(a0.v, b0.v, acc[j]);
      b1.q[0] = *(const uint4*)&Bb[rn + 32 + hi * 8];
      b1.q[1] = *(const uint4*)&Bb[rn + 48 + hi * 8];
      acc[j] = wmma_bf16(a1.v, b1.v, acc[j]);
    }
  }

  const int col0 = n0 + nw * 64;
#pragma unroll
  for (int j = 0; j < 4; ++j) {
    const int col = col0 + j * 16 + lh;
    const float bv = bias ? bias[col] : 0.0f;
#pragma unroll
    for (int r = 0; r < 8; ++r) {
      const int row = m0 + mw * 16 + r + 8 * hi; // C/D layout: VGPR r -> M=r / r+8
      float vv = acc[j][r] + bv;
      if (do_gelu) vv = gelu_f(vv);
      if (resid) vv += resid[(size_t)row * N + col];
      if (out_f32)  out_f32[(size_t)row * N + col]  = vv;
      if (out_bf16) out_bf16[(size_t)row * N + col] = (__bf16)vv;
    }
  }
}

// ---------------------------------------------------------------------------
// Causal flash attention. qkv: [B*T, 3C] bf16 rows (Q | K | V per row, head h
// at col h*64). Each wave owns one 16-row q-tile, iterates 32-key blocks with
// online softmax; per-wave private LDS (no WG barriers: divergent causal trip
// counts). V tile DMA'd async to LDS, overlapping the QK^T WMMAs + softmax.
// ---------------------------------------------------------------------------
__global__ __launch_bounds__(256) void attention_kernel(
    const __bf16* __restrict__ qkv, __bf16* __restrict__ outp) {
  const int bh = blockIdx.x;
  const int b = bh / NH_, h = bh % NH_;
  const int w    = threadIdx.x >> 5;
  const int lane = threadIdx.x & 31;
  const int lh = lane & 15, hi = lane >> 4;
  const int q0 = blockIdx.y * 128 + w * 16;

  __shared__ __bf16 smem[8 * 2560]; // per wave: V tile 32x64 (2048) + P 16x32 (512)
  __bf16* Vs = &smem[w * 2560];
  __bf16* Ps = &smem[w * 2560 + 2048];
  const uint32_t vs_base = lds_lo(&Vs[lane * 64]);

  const int rowstride = 3 * C_;
  const __bf16* qbase = qkv + (size_t)(b * T_) * rowstride + h * HD_;
  const __bf16* kptr  = qbase + C_;
  const __bf16* vptr  = qbase + 2 * C_;

  // Q fragments: 16 rows x 64 head-dim -> two 16x32 A-frags (held in regs)
  FragB16 qf[2];
  {
    const __bf16* qrow = qbase + (size_t)(q0 + lh) * rowstride;
#pragma unroll
    for (int f = 0; f < 2; ++f) {
      qf[f].q[0] = *(const uint4*)(qrow + f * 32 + hi * 8);
      qf[f].q[1] = *(const uint4*)(qrow + f * 32 + 16 + hi * 8);
    }
  }

  v8f o[4];
  float mstate[8], lstate[8];
#pragma unroll
  for (int j = 0; j < 4; ++j)
#pragma unroll
    for (int e = 0; e < 8; ++e) o[j][e] = 0.0f;
#pragma unroll
  for (int r = 0; r < 8; ++r) { mstate[r] = -1e30f; lstate[r] = 0.0f; }

  const int nkb = (q0 + 47) >> 5; // key blocks overlapping causal window
  for (int kb = 0; kb < nkb; ++kb) {
    const int kbase = kb * 32;

    // Kick off async DMA of V[32 keys x 64 hd] into per-wave LDS.
    // Prior block's LDS reads are complete (wait) before DMA may overwrite.
    wait_ds0();
    {
      const __bf16* vrow = vptr + (size_t)(kbase + lane) * rowstride;
#pragma unroll
      for (int c = 0; c < 8; ++c)
        async_ld_b128(vs_base + c * 16, vrow + c * 8);
    }

    // S = Q * K^T (two 16-key subtiles, K-dim 64 = 2 wmma each) - overlaps DMA
    v8f s[2];
#pragma unroll
    for (int j = 0; j < 2; ++j) {
#pragma unroll
      for (int e = 0; e < 8; ++e) s[j][e] = 0.0f;
      const __bf16* krow = kptr + (size_t)(kbase + j * 16 + lh) * rowstride;
      FragB16 kf0, kf1;
      kf0.q[0] = *(const uint4*)(krow + hi * 8);
      kf0.q[1] = *(const uint4*)(krow + 16 + hi * 8);
      kf1.q[0] = *(const uint4*)(krow + 32 + hi * 8);
      kf1.q[1] = *(const uint4*)(krow + 48 + hi * 8);
      s[j] = wmma_bf16(qf[0].v, kf0.v, s[j]);
      s[j] = wmma_bf16(qf[1].v, kf1.v, s[j]);
    }

    // Scale, causal mask (select: EXEC stays all-1s), row max via xor-shuffle
    float rmax[8], alpha[8];
#pragma unroll
    for (int r = 0; r < 8; ++r) {
      const int qrow = q0 + r + 8 * hi;
      float m = -1e30f;
#pragma unroll
      for (int j = 0; j < 2; ++j) {
        const int key = kbase + j * 16 + lh;
        float sv = s[j][r] * 0.125f; // 1/sqrt(64)
        sv = (key <= qrow) ? sv : -1e30f;
        s[j][r] = sv;
        m = fmaxf(m, sv);
      }
#pragma unroll
      for (int off = 1; off < 16; off <<= 1)
        m = fmaxf(m, __shfl_xor(m, off, 32)); // stays within 16-lane halves
      rmax[r] = m;
    }
    // Online softmax update
#pragma unroll
    for (int r = 0; r < 8; ++r) {
      const float mn = fmaxf(mstate[r], rmax[r]);
      alpha[r] = __expf(mstate[r] - mn);
      mstate[r] = mn;
      float psum = 0.0f;
#pragma unroll
      for (int j = 0; j < 2; ++j) {
        const float p = __expf(s[j][r] - mn);
        s[j][r] = p;
        psum += p;
      }
#pragma unroll
      for (int off = 1; off < 16; off <<= 1)
        psum += __shfl_xor(psum, off, 32);
      lstate[r] = lstate[r] * alpha[r] + psum;
    }

    // P (C/D layout) -> LDS bf16 -> reload in A-fragment layout
#pragma unroll
    for (int r = 0; r < 8; ++r) {
      const int prow = r + 8 * hi;
      Ps[prow * 32 + lh]      = (__bf16)s[0][r];
      Ps[prow * 32 + 16 + lh] = (__bf16)s[1][r];
    }
    wait_ds0(); // P stores visible to this wave's loads
    FragB16 pf;
    pf.q[0] = *(const uint4*)&Ps[lh * 32 + hi * 8];
    pf.q[1] = *(const uint4*)&Ps[lh * 32 + 16 + hi * 8];

    wait_async0(); // V tile DMA complete

    // O = O*alpha + P*V (4 hd subtiles); V B-frag gathered from LDS
#pragma unroll
    for (int jj = 0; jj < 4; ++jj) {
#pragma unroll
      for (int r = 0; r < 8; ++r) o[jj][r] *= alpha[r];
      FragB16 vf;
      const int col = jj * 16 + lh;
#pragma unroll
      for (int e = 0; e < 8; ++e) {
        vf.v[e]     = Vs[(hi * 8 + e) * 64 + col];      // K = hi*8 .. +7
        vf.v[8 + e] = Vs[(16 + hi * 8 + e) * 64 + col]; // K = 16+hi*8 ..
      }
      o[jj] = wmma_bf16(pf.v, vf.v, o[jj]);
    }
  }

  // Normalize and store [B,T,C] bf16
#pragma unroll
  for (int jj = 0; jj < 4; ++jj) {
#pragma unroll
    for (int r = 0; r < 8; ++r) {
      const int row = q0 + r + 8 * hi;
      const float val = o[jj][r] / lstate[r];
      outp[(size_t)(b * T_ + row) * C_ + h * HD_ + jj * 16 + lh] = (__bf16)val;
    }
  }
}

// ---------------------------------------------------------------------------
// Host-side orchestration
// ---------------------------------------------------------------------------
extern "C" void kernel_launch(void* const* d_in, const int* in_sizes, int n_in,
                              void* d_out, int out_size, void* d_ws, size_t ws_size,
                              hipStream_t stream) {
  (void)in_sizes; (void)n_in; (void)out_size; (void)ws_size;
  const float* x     = (const float*)d_in[0];
  const float* ln1_g = (const float*)d_in[1];
  const float* ln1_b = (const float*)d_in[2];
  const float* w_qkv = (const float*)d_in[3];
  const float* b_qkv = (const float*)d_in[4];
  const float* w_ap  = (const float*)d_in[5];
  const float* b_ap  = (const float*)d_in[6];
  const float* ln2_g = (const float*)d_in[7];
  const float* ln2_b = (const float*)d_in[8];
  const float* w_fc  = (const float*)d_in[9];
  const float* b_fc  = (const float*)d_in[10];
  const float* w_mp  = (const float*)d_in[11];
  const float* b_mp  = (const float*)d_in[12];
  float* out = (float*)d_out;

  char* ws = (char*)d_ws;
  size_t off = 0;
  auto take = [&](size_t bytes) {
    char* p = ws + off;
    off = (off + bytes + 255) & ~(size_t)255;
    return p;
  };
  __bf16* wqkvT = (__bf16*)take((size_t)3 * C_ * C_ * 2); // [3C, C]
  __bf16* wapT  = (__bf16*)take((size_t)C_ * C_ * 2);     // [C, C]
  __bf16* wfcT  = (__bf16*)take((size_t)FF_ * C_ * 2);    // [FF, C]
  __bf16* wmpT  = (__bf16*)take((size_t)C_ * FF_ * 2);    // [C, FF]
  __bf16* h1    = (__bf16*)take((size_t)BT_ * C_ * 2);
  __bf16* qkvb  = (__bf16*)take((size_t)BT_ * 3 * C_ * 2);
  __bf16* attnb = (__bf16*)take((size_t)BT_ * C_ * 2);
  float*  x1    = (float*) take((size_t)BT_ * C_ * 4);
  __bf16* h2    = (__bf16*)take((size_t)BT_ * C_ * 2);
  __bf16* hf    = (__bf16*)take((size_t)BT_ * FF_ * 2);

  const dim3 tb(32, 8);
  transpose_convert_kernel<<<dim3(3 * C_ / 32, C_ / 32), tb, 0, stream>>>(w_qkv, wqkvT, C_, 3 * C_);
  transpose_convert_kernel<<<dim3(C_ / 32, C_ / 32),     tb, 0, stream>>>(w_ap,  wapT,  C_, C_);
  transpose_convert_kernel<<<dim3(FF_ / 32, C_ / 32),    tb, 0, stream>>>(w_fc,  wfcT,  C_, FF_);
  transpose_convert_kernel<<<dim3(C_ / 32, FF_ / 32),    tb, 0, stream>>>(w_mp,  wmpT,  FF_, C_);

  // h1 = LN1(x)
  layernorm_bf16_kernel<<<BT_, 256, 0, stream>>>(x, ln1_g, ln1_b, h1);
  // qkv = h1 @ w_qkv + b_qkv
  gemm_bf16_kernel<<<dim3(3 * C_ / 128, BT_ / 64), 256, 0, stream>>>(
      h1, wqkvT, b_qkv, nullptr, nullptr, qkvb, BT_, 3 * C_, C_, 0);
  // attn = causal flash attention
  attention_kernel<<<dim3(B_ * NH_, T_ / 128), 256, 0, stream>>>(qkvb, attnb);
  // x1 = x + attn @ w_attn_proj + b_attn_proj
  gemm_bf16_kernel<<<dim3(C_ / 128, BT_ / 64), 256, 0, stream>>>(
      attnb, wapT, b_ap, x, x1, nullptr, BT_, C_, C_, 0);
  // h2 = LN2(x1)
  layernorm_bf16_kernel<<<BT_, 256, 0, stream>>>(x1, ln2_g, ln2_b, h2);
  // hf = gelu(h2 @ w_fc + b_fc)
  gemm_bf16_kernel<<<dim3(FF_ / 128, BT_ / 64), 256, 0, stream>>>(
      h2, wfcT, b_fc, nullptr, nullptr, hf, BT_, FF_, C_, 1);
  // out = x1 + hf @ w_mlp_proj + b_mlp_proj
  gemm_bf16_kernel<<<dim3(C_ / 128, BT_ / 64), 256, 0, stream>>>(
      hf, wmpT, b_mp, x1, out, nullptr, BT_, C_, FF_, 0);
}
